// Struct2SeqGCN_87832081203765
// MI455X (gfx1250) — matrix operand order repaired and checked by reading, not compiled, not run on hardware
//
#include <hip/hip_runtime.h>
#include <hip/hip_bf16.h>

typedef __attribute__((ext_vector_type(16))) _Float16 v16h;
typedef __attribute__((ext_vector_type(8)))  _Float16 v8h;
typedef __attribute__((ext_vector_type(8)))  float    v8f;

#define WSTRIDE 136   // padded f16 column stride: 272B -> conflict-free half-wave

// Build a 16-f16 WMMA fragment from two 16B-aligned LDS chunks.
__device__ __forceinline__ v16h load_frag(const _Float16* p)
{
    const v8h* bp = (const v8h*)p;
    v8h lo = bp[0];
    v8h hi = bp[1];
    return __builtin_shufflevector(lo, hi, 0, 1, 2, 3, 4, 5, 6, 7,
                                           8, 9, 10, 11, 12, 13, 14, 15);
}

// Build the A fragment (16x32 f16, contiguous k, same order as B) for one K-tile:
// lane needs k = kbase+0..15 where kbase = kt*32 + 16*hi.
__device__ __forceinline__ v16h load_a_frag(const float* __restrict__ xrow,
                                            int kt, int hi)
{
    const float4* p0 = (const float4*)(xrow + kt * 32 + 16 * hi);
    const float4* p1 = (const float4*)(xrow + kt * 32 + 8 + 16 * hi);
    const float4 c0 = p0[0], c1 = p0[1];
    const float4 c2 = p1[0], c3 = p1[1];
    v16h a;
    a[0]  = (_Float16)c0.x; a[1]  = (_Float16)c0.y;
    a[2]  = (_Float16)c0.z; a[3]  = (_Float16)c0.w;
    a[4]  = (_Float16)c1.x; a[5]  = (_Float16)c1.y;
    a[6]  = (_Float16)c1.z; a[7]  = (_Float16)c1.w;
    a[8]  = (_Float16)c2.x; a[9]  = (_Float16)c2.y;
    a[10] = (_Float16)c2.z; a[11] = (_Float16)c2.w;
    a[12] = (_Float16)c3.x; a[13] = (_Float16)c3.y;
    a[14] = (_Float16)c3.z; a[15] = (_Float16)c3.w;
    return a;
}

// ---------------------------------------------------------------------------
// Dense GEMM  H[N,128] = X[N,128] @ W[128,128]   (f16 WMMA, f32 accumulate)
// Block = 256 threads = 8 wave32; each wave computes a 16x128 output stripe.
// W staged in LDS transposed (Wp[col][k]); all A fragments preloaded, B
// fragments batch-loaded per K-tile so DS latency overlaps the WMMA burst.
// ---------------------------------------------------------------------------
__global__ __launch_bounds__(256) void gemm128_wmma(
    const float* __restrict__ X, const float* __restrict__ W,
    float* __restrict__ H, int n_rows)
{
    __shared__ _Float16 Wp[128 * WSTRIDE];         // ~34 KB of 320 KB WGP LDS
    for (int i = threadIdx.x; i < 128 * 128; i += 256) {
        const int k = i >> 7, col = i & 127;
        Wp[col * WSTRIDE + k] = (_Float16)W[i];
    }
    __syncthreads();

    const int wave = threadIdx.x >> 5;
    const int lane = threadIdx.x & 31;
    const int hi   = lane >> 4;                    // half-wave select
    const int lo   = lane & 15;
    const int m0   = (blockIdx.x * 8 + wave) * 16;
    if (m0 >= n_rows) return;

    const int mr = (m0 + lo < n_rows) ? (m0 + lo) : (n_rows - 1);
    const float* __restrict__ xrow = X + (size_t)mr * 128;

    // Preload all four A fragments (16 global_load_b128 in flight).
    v16h afrag[4];
    #pragma unroll
    for (int kt = 0; kt < 4; ++kt)
        afrag[kt] = load_a_frag(xrow, kt, hi);

    v8f acc[8];
    #pragma unroll
    for (int nt = 0; nt < 8; ++nt)
        #pragma unroll
        for (int j = 0; j < 8; ++j) acc[nt][j] = 0.0f;

    #pragma unroll
    for (int kt = 0; kt < 4; ++kt) {
        // Batch-load all 8 B fragments for this K-tile (16 ds_load_b128),
        // then fire 8 independent WMMAs.
        v16h bfrag[8];
        #pragma unroll
        for (int nt = 0; nt < 8; ++nt)
            bfrag[nt] = load_frag(&Wp[(nt * 16 + lo) * WSTRIDE + kt * 32 + 16 * hi]);
        #pragma unroll
        for (int nt = 0; nt < 8; ++nt)
            acc[nt] = __builtin_amdgcn_wmma_f32_16x16x32_f16(
                false, afrag[kt], false, bfrag[nt], (short)0, acc[nt],
                false, false);
    }

    #pragma unroll
    for (int nt = 0; nt < 8; ++nt) {
        const int col = nt * 16 + lo;
        #pragma unroll
        for (int r = 0; r < 8; ++r) {
            const int row = m0 + r + 8 * hi;       // D layout: VGPR r -> M=r+8*hi
            if (row < n_rows)
                H[(size_t)row * 128 + col] = acc[nt][r];
        }
    }
}

// ---------------------------------------------------------------------------
// FC head: out[N,21] = X[N,128] @ Wfc[128,21] + bfc   (padded to 32 cols)
// ---------------------------------------------------------------------------
__global__ __launch_bounds__(256) void fc_wmma(
    const float* __restrict__ X, const float* __restrict__ Wfc,
    const float* __restrict__ bfc, float* __restrict__ out,
    int n_rows, int n_cls)
{
    __shared__ _Float16 Wp[32 * WSTRIDE];          // ~8.5 KB, zero-padded cols
    for (int i = threadIdx.x; i < 32 * 128; i += 256) {
        const int col = i >> 7, k = i & 127;
        Wp[col * WSTRIDE + k] =
            (_Float16)((col < n_cls) ? Wfc[k * n_cls + col] : 0.0f);
    }
    __syncthreads();

    const int wave = threadIdx.x >> 5;
    const int lane = threadIdx.x & 31;
    const int hi   = lane >> 4;
    const int lo   = lane & 15;
    const int m0   = (blockIdx.x * 8 + wave) * 16;
    if (m0 >= n_rows) return;

    const int mr = (m0 + lo < n_rows) ? (m0 + lo) : (n_rows - 1);
    const float* __restrict__ xrow = X + (size_t)mr * 128;

    v16h afrag[4];
    #pragma unroll
    for (int kt = 0; kt < 4; ++kt)
        afrag[kt] = load_a_frag(xrow, kt, hi);

    v8f acc[2];
    #pragma unroll
    for (int nt = 0; nt < 2; ++nt)
        #pragma unroll
        for (int j = 0; j < 8; ++j) acc[nt][j] = 0.0f;

    #pragma unroll
    for (int kt = 0; kt < 4; ++kt) {
        v16h bfrag[2];
        #pragma unroll
        for (int nt = 0; nt < 2; ++nt)
            bfrag[nt] = load_frag(&Wp[(nt * 16 + lo) * WSTRIDE + kt * 32 + 16 * hi]);
        #pragma unroll
        for (int nt = 0; nt < 2; ++nt)
            acc[nt] = __builtin_amdgcn_wmma_f32_16x16x32_f16(
                false, afrag[kt], false, bfrag[nt], (short)0, acc[nt],
                false, false);
    }

    #pragma unroll
    for (int nt = 0; nt < 2; ++nt) {
        const int col = nt * 16 + lo;
        if (col >= n_cls) continue;
        const float bias = bfc[col];
        #pragma unroll
        for (int r = 0; r < 8; ++r) {
            const int row = m0 + r + 8 * hi;
            if (row < n_rows)
                out[(size_t)row * n_cls + col] = acc[nt][r] + bias;
        }
    }
}

// ---------------------------------------------------------------------------
// Helpers: zero, degree count, rsqrt, edge scatter, finalize(+bias,+ReLU)
// ---------------------------------------------------------------------------
__global__ void zero_kernel(float* __restrict__ p, long long n)
{
    long long i = (long long)blockIdx.x * blockDim.x + threadIdx.x;
    const long long stride = (long long)gridDim.x * blockDim.x;
    for (; i < n; i += stride) p[i] = 0.0f;
}

__global__ void deg_count_kernel(const long long* __restrict__ dst,
                                 float* __restrict__ deg, int n_edges)
{
    const int i = blockIdx.x * blockDim.x + threadIdx.x;
    if (i < n_edges) atomicAdd(&deg[dst[i]], 1.0f);
}

__global__ void deg_rsqrt_kernel(float* __restrict__ deg, int n)
{
    const int i = blockIdx.x * blockDim.x + threadIdx.x;
    if (i < n) deg[i] = rsqrtf(deg[i] + 1.0f);
}

// One wave32 per edge; lane handles 4 consecutive features (float4).
__global__ __launch_bounds__(256) void scatter_edges_kernel(
    const float* __restrict__ H, const long long* __restrict__ src,
    const long long* __restrict__ dst, const float* __restrict__ dinv,
    float* __restrict__ agg, int n_edges)
{
    const int e = blockIdx.x * 8 + (threadIdx.x >> 5);
    if (e >= n_edges) return;
    const int lane = threadIdx.x & 31;
    const long long s = src[e];
    const long long d = dst[e];
    __builtin_prefetch(H + (size_t)s * 128 + lane * 4, 0, 0);   // global_prefetch_b8
    const float norm = dinv[s] * dinv[d];
    const float4 v = ((const float4*)(H + (size_t)s * 128))[lane];
    float* ap = agg + (size_t)d * 128 + lane * 4;
    atomicAdd(ap + 0, v.x * norm);
    atomicAdd(ap + 1, v.y * norm);
    atomicAdd(ap + 2, v.z * norm);
    atomicAdd(ap + 3, v.w * norm);
}

// agg = relu(agg + h * dinv^2 + bias[col])   (in place on agg)
__global__ void finalize_kernel(float* __restrict__ agg,
                                const float* __restrict__ h,
                                const float* __restrict__ dinv,
                                const float* __restrict__ bias, long long n)
{
    long long i = (long long)blockIdx.x * blockDim.x + threadIdx.x;
    const long long stride = (long long)gridDim.x * blockDim.x;
    for (; i < n; i += stride) {
        const int row = (int)(i >> 7);
        const int col = (int)(i & 127);
        const float d = dinv[row];
        const float v = agg[i] + h[i] * d * d + bias[col];
        agg[i] = fmaxf(v, 0.0f);
    }
}

// ---------------------------------------------------------------------------
// Orchestration
// ---------------------------------------------------------------------------
extern "C" void kernel_launch(void* const* d_in, const int* in_sizes, int n_in,
                              void* d_out, int out_size, void* d_ws, size_t ws_size,
                              hipStream_t stream)
{
    const float*     x    = (const float*)d_in[0];
    const long long* ei   = (const long long*)d_in[1];
    const float*     W1   = (const float*)d_in[2];
    const float*     b1   = (const float*)d_in[3];
    const float*     W2   = (const float*)d_in[4];
    const float*     b2   = (const float*)d_in[5];
    const float*     Wfc  = (const float*)d_in[6];
    const float*     bfc  = (const float*)d_in[7];
    float*           out  = (float*)d_out;

    const int N     = in_sizes[0] / 128;
    const int E     = in_sizes[1] / 2;
    const int n_cls = in_sizes[7];
    const long long NF = (long long)N * 128;

    // Workspace layout: dinv [N] | B1 [N*128] | B2 [N*128]  (~51.6 MB)
    char*  ws   = (char*)d_ws;
    float* dinv = (float*)ws;
    size_t off  = (((size_t)N * sizeof(float)) + 255) & ~(size_t)255;
    float* B1   = (float*)(ws + off);
    float* B2   = B1 + (size_t)NF;

    const long long* src = ei;
    const long long* dst = ei + E;

    const int gemm_blocks = (N + 127) / 128;
    const int zgridN  = (N + 255) / 256;
    const int zgridNF = (int)((NF + 255) / 256);
    const int egrid   = (E + 255) / 256;
    const int sgrid   = (E + 7) / 8;

    // degrees -> dinv = rsqrt(deg + 1)
    zero_kernel<<<zgridN, 256, 0, stream>>>(dinv, N);
    deg_count_kernel<<<egrid, 256, 0, stream>>>(dst, dinv, E);
    deg_rsqrt_kernel<<<zgridN, 256, 0, stream>>>(dinv, N);

    // Layer 1: B2 = relu(Agg(x@W1) + (x@W1)*d^2 + b1)
    gemm128_wmma<<<gemm_blocks, 256, 0, stream>>>(x, W1, B1, N);
    zero_kernel<<<zgridNF, 256, 0, stream>>>(B2, NF);
    scatter_edges_kernel<<<sgrid, 256, 0, stream>>>(B1, src, dst, dinv, B2, E);
    finalize_kernel<<<zgridNF, 256, 0, stream>>>(B2, B1, dinv, b1, NF);

    // Layer 2: B2 = relu(Agg(B2@W2) + (B2@W2)*d^2 + b2)
    gemm128_wmma<<<gemm_blocks, 256, 0, stream>>>(B2, W2, B1, N);
    zero_kernel<<<zgridNF, 256, 0, stream>>>(B2, NF);  // runs after gemm consumed B2
    scatter_edges_kernel<<<sgrid, 256, 0, stream>>>(B1, src, dst, dinv, B2, E);
    finalize_kernel<<<zgridNF, 256, 0, stream>>>(B2, B1, dinv, b2, NF);

    // FC head
    fc_wmma<<<gemm_blocks, 256, 0, stream>>>(B2, Wfc, bfc, out, N, n_cls);
}
